// DT4Rec_75342316306617
// MI455X (gfx1250) — matrix operand order repaired and checked by reading, not compiled, not run on hardware
//
#include <hip/hip_runtime.h>
#include <hip/hip_bf16.h>
#include <math.h>

typedef _Float16 v4h  __attribute__((ext_vector_type(4)));
typedef _Float16 v8h  __attribute__((ext_vector_type(8)));
typedef _Float16 v16h __attribute__((ext_vector_type(16)));
typedef float    v8f  __attribute__((ext_vector_type(8)));

__device__ __forceinline__ float gelu_exact(float x) {
    return 0.5f * x * (1.0f + erff(x * 0.7071067811865475f));
}
__device__ __forceinline__ float f4c(const float4& f, int n) {
    return (&f.x)[n];
}

// ---------------------------------------------------------------------------
// FAST WMMA GEMM: C[M,N] = act(A[M,K] @ W[K,N] + bias) (+ residual)
// Requires M%128==0, N%128==0, K%32==0 (true for all big GEMMs here).
// Block tile 128x128x32, 256 threads = 8 wave32 waves (4x2), each wave owns
// 32x64 via 2x4 v_wmma_f32_16x16x32_f16 accumulators. Double-buffered LDS:
// global b128 loads of tile kt+1 are issued before the WMMAs of tile kt.
// ---------------------------------------------------------------------------
#define FBM 128
#define FBN 128
#define FBK 32
#define LDA 40   // padded LDS row stride (halves); 80B keeps 16B alignment
#define LDB 40

template <int ACT, bool RES>
__global__ __launch_bounds__(256) void gemm_wmma_fast(
    const float* __restrict__ A, const float* __restrict__ W,
    const float* __restrict__ bias, const float* __restrict__ res,
    float* __restrict__ C, int M, int N, int K)
{
    __shared__ __align__(16) _Float16 As[2][FBM * LDA];   // [m][k]
    __shared__ __align__(16) _Float16 Bs[2][FBN * LDB];   // transposed [n][k]

    const int tid  = threadIdx.x;
    const int lane = tid & 31;
    const int wave = tid >> 5;
    const int wm   = wave >> 1;      // 0..3 -> 32-row strip
    const int wn   = wave & 1;       // 0..1 -> 64-col strip
    const int m0b  = blockIdx.y * FBM;
    const int n0b  = blockIdx.x * FBN;

    // B staging: each thread owns a 4(k) x 4(n) micro-block.
    const int kgB = (tid >> 5) << 2;     // k group: 0,4,..,28
    const int nB  = (tid & 31) << 2;     // n offset: 0,4,..,124

    v8f acc[2][4] = {};
    float4 fa[4], fb[4];

    const int nk = K / FBK;

    // ---- prologue: load + stage tile 0 ----
#pragma unroll
    for (int i = 0; i < 4; ++i) {
        int idx = tid + i * 256;
        int r = idx >> 3, c4 = (idx & 7) << 2;
        fa[i] = *(const float4*)(A + (size_t)(m0b + r) * K + c4);
    }
#pragma unroll
    for (int j = 0; j < 4; ++j)
        fb[j] = *(const float4*)(W + (size_t)(kgB + j) * N + n0b + nB);
#pragma unroll
    for (int i = 0; i < 4; ++i) {
        int idx = tid + i * 256;
        int r = idx >> 3, c4 = (idx & 7) << 2;
        v4h h = { (_Float16)fa[i].x, (_Float16)fa[i].y,
                  (_Float16)fa[i].z, (_Float16)fa[i].w };
        *(v4h*)&As[0][r * LDA + c4] = h;
    }
#pragma unroll
    for (int n = 0; n < 4; ++n) {
        v4h h = { (_Float16)f4c(fb[0], n), (_Float16)f4c(fb[1], n),
                  (_Float16)f4c(fb[2], n), (_Float16)f4c(fb[3], n) };
        *(v4h*)&Bs[0][(nB + n) * LDB + kgB] = h;
    }
    __syncthreads();

    for (int kt = 0; kt < nk; ++kt) {
        const int cur = kt & 1;
        const int nxt = cur ^ 1;

        // ---- issue global loads for tile kt+1 (no wait yet) ----
        if (kt + 1 < nk) {
            const int k1 = (kt + 1) * FBK;
#pragma unroll
            for (int i = 0; i < 4; ++i) {
                int idx = tid + i * 256;
                int r = idx >> 3, c4 = (idx & 7) << 2;
                fa[i] = *(const float4*)(A + (size_t)(m0b + r) * K + k1 + c4);
            }
#pragma unroll
            for (int j = 0; j < 4; ++j)
                fb[j] = *(const float4*)(W + (size_t)(k1 + kgB + j) * N + n0b + nB);
        }

        // ---- fragments (ISA 16-bit A/B lane layouts) + 8 WMMAs ----
        v16h afrag[2], bfrag[4];
        {
            const int r  = lane & 15;
            const int kb = (lane >> 4) << 3;   // 0 or 8
#pragma unroll
            for (int f = 0; f < 2; ++f) {
                const _Float16* p = &As[cur][(wm * 32 + f * 16 + r) * LDA + kb];
                v8h lo = *(const v8h*)p;         // K = kb .. kb+7
                v8h hi = *(const v8h*)(p + 16);  // K = kb+16 .. kb+23
                afrag[f] = __builtin_shufflevector(lo, hi,
                    0,1,2,3,4,5,6,7,8,9,10,11,12,13,14,15);
            }
            const int nn  = lane & 15;
            const int kb2 = (lane >> 4) << 3;  // 0 or 8
#pragma unroll
            for (int g = 0; g < 4; ++g) {
                const _Float16* p = &Bs[cur][(wn * 64 + g * 16 + nn) * LDB + kb2];
                v8h lo = *(const v8h*)p;        // K = kb2 .. kb2+7
                v8h hi = *(const v8h*)(p + 16); // K = kb2+16 .. kb2+23
                bfrag[g] = __builtin_shufflevector(lo, hi,
                    0,1,2,3,4,5,6,7,8,9,10,11,12,13,14,15);
            }
        }
#pragma unroll
        for (int f = 0; f < 2; ++f)
#pragma unroll
            for (int g = 0; g < 4; ++g)
                acc[f][g] = __builtin_amdgcn_wmma_f32_16x16x32_f16(
                    false, afrag[f], false, bfrag[g],
                    (short)0, acc[f][g], false, false);

        // ---- convert + stage tile kt+1 into the other buffer ----
        if (kt + 1 < nk) {
#pragma unroll
            for (int i = 0; i < 4; ++i) {
                int idx = tid + i * 256;
                int r = idx >> 3, c4 = (idx & 7) << 2;
                v4h h = { (_Float16)fa[i].x, (_Float16)fa[i].y,
                          (_Float16)fa[i].z, (_Float16)fa[i].w };
                *(v4h*)&As[nxt][r * LDA + c4] = h;
            }
#pragma unroll
            for (int n = 0; n < 4; ++n) {
                v4h h = { (_Float16)f4c(fb[0], n), (_Float16)f4c(fb[1], n),
                          (_Float16)f4c(fb[2], n), (_Float16)f4c(fb[3], n) };
                *(v4h*)&Bs[nxt][(nB + n) * LDB + kgB] = h;
            }
        }
        __syncthreads();
    }

    // --- epilogue: lanes 0-15: M=r, N=lane; lanes 16-31: M=8+r, N=lane-16 ---
    const int ncol = lane & 15;
    const int mofs = (lane >> 4) * 8;
#pragma unroll
    for (int f = 0; f < 2; ++f) {
#pragma unroll
        for (int g = 0; g < 4; ++g) {
            const int gn = n0b + wn * 64 + g * 16 + ncol;
            const float bv = bias[gn];
#pragma unroll
            for (int r = 0; r < 8; ++r) {
                const int gm = m0b + wm * 32 + f * 16 + mofs + r;
                float v = acc[f][g][r] + bv;
                if (ACT == 1) v = gelu_exact(v);
                if (RES) v += res[(size_t)gm * N + gn];
                C[(size_t)gm * N + gn] = v;
            }
        }
    }
}

// ---------------------------------------------------------------------------
// EDGE WMMA GEMM (guarded): for K=91 / K=27 / N=27 cases only.
// ---------------------------------------------------------------------------
#define BM 128
#define BN 64
#define BK 32

template <int ACT, bool RES>
__global__ __launch_bounds__(256) void gemm_wmma_edge(
    const float* __restrict__ A, const float* __restrict__ W,
    const float* __restrict__ bias, const float* __restrict__ res,
    float* __restrict__ C, int M, int N, int K)
{
    __shared__ __align__(16) _Float16 As[BM * BK];
    __shared__ __align__(16) _Float16 Bs[BN * BK];   // transposed [n][k]

    const int tid  = threadIdx.x;
    const int lane = tid & 31;
    const int wave = tid >> 5;
    const int wm   = wave >> 1;
    const int wn   = wave & 1;
    const int m0b  = blockIdx.y * BM;
    const int n0b  = blockIdx.x * BN;

    v8f acc[2][2] = {};

    const int nk = (K + BK - 1) / BK;
    for (int kt = 0; kt < nk; ++kt) {
        const int k0 = kt * BK;
#pragma unroll
        for (int i = 0; i < 16; ++i) {
            int idx = tid + i * 256;
            int r = idx >> 5, c = idx & 31;
            int gm = m0b + r, gk = k0 + c;
            float v = (gm < M && gk < K) ? A[(size_t)gm * K + gk] : 0.0f;
            As[r * BK + c] = (_Float16)v;
        }
#pragma unroll
        for (int i = 0; i < 8; ++i) {
            int idx = tid + i * 256;
            int kk = idx >> 6, nn = idx & 63;
            int gk = k0 + kk, gn = n0b + nn;
            float v = (gk < K && gn < N) ? W[(size_t)gk * N + gn] : 0.0f;
            Bs[nn * BK + kk] = (_Float16)v;
        }
        __syncthreads();

        v16h afrag[2], bfrag[2];
        {
            const int r  = lane & 15;
            const int kb = (lane >> 4) << 3;
#pragma unroll
            for (int f = 0; f < 2; ++f) {
                const _Float16* p = &As[(wm * 32 + f * 16 + r) * BK + kb];
                v8h lo = *(const v8h*)p;
                v8h hi = *(const v8h*)(p + 16);
                afrag[f] = __builtin_shufflevector(lo, hi,
                    0,1,2,3,4,5,6,7,8,9,10,11,12,13,14,15);
            }
            const int nn  = lane & 15;
            const int kb2 = (lane >> 4) << 3;
#pragma unroll
            for (int g = 0; g < 2; ++g) {
                const _Float16* p = &Bs[(wn * 32 + g * 16 + nn) * BK + kb2];
                v8h lo = *(const v8h*)p;
                v8h hi = *(const v8h*)(p + 16);
                bfrag[g] = __builtin_shufflevector(lo, hi,
                    0,1,2,3,4,5,6,7,8,9,10,11,12,13,14,15);
            }
        }
#pragma unroll
        for (int f = 0; f < 2; ++f)
#pragma unroll
            for (int g = 0; g < 2; ++g)
                acc[f][g] = __builtin_amdgcn_wmma_f32_16x16x32_f16(
                    false, afrag[f], false, bfrag[g],
                    (short)0, acc[f][g], false, false);
        __syncthreads();
    }

    const int ncol = lane & 15;
    const int mofs = (lane >> 4) * 8;
#pragma unroll
    for (int f = 0; f < 2; ++f) {
#pragma unroll
        for (int g = 0; g < 2; ++g) {
#pragma unroll
            for (int r = 0; r < 8; ++r) {
                int gm = m0b + wm * 32 + f * 16 + mofs + r;
                int gn = n0b + wn * 32 + g * 16 + ncol;
                if (gm < M && gn < N) {
                    float v = acc[f][g][r];
                    if (bias) v += bias[gn];
                    if (ACT == 1) v = gelu_exact(v);
                    if (RES) v += res[(size_t)gm * N + gn];
                    C[(size_t)gm * N + gn] = v;
                }
            }
        }
    }
}

// ---------------------------------------------------------------------------
// LayerNorm over D=1024, one block per row.
// ---------------------------------------------------------------------------
__global__ __launch_bounds__(256) void ln_kernel(
    const float* __restrict__ X, const float* __restrict__ g,
    const float* __restrict__ b, float* __restrict__ O, int D)
{
    __shared__ float rs[256], rq[256];
    const int row = blockIdx.x, tid = threadIdx.x;
    const float* x = X + (size_t)row * D;
    float s = 0.f, s2 = 0.f;
    for (int d = tid; d < D; d += 256) { float v = x[d]; s += v; s2 += v * v; }
    rs[tid] = s; rq[tid] = s2; __syncthreads();
    for (int off = 128; off > 0; off >>= 1) {
        if (tid < off) { rs[tid] += rs[tid + off]; rq[tid] += rq[tid + off]; }
        __syncthreads();
    }
    float mu  = rs[0] / D;
    float var = rq[0] / D - mu * mu;
    float inv = rsqrtf(var + 1e-5f);
    for (int d = tid; d < D; d += 256)
        O[(size_t)row * D + d] = (x[d] - mu) * inv * g[d] + b[d];
}

// ---------------------------------------------------------------------------
// Fused causal attention, one block per (b, h). T=192, hd=64, mask==ones.
// ---------------------------------------------------------------------------
__global__ __launch_bounds__(192) void attn_kernel(
    const float* __restrict__ Q, const float* __restrict__ Kx,
    const float* __restrict__ Vx, float* __restrict__ Y)
{
    const int T = 192, HD = 64, Dd = 1024;
    const int bh = blockIdx.x, b = bh >> 4, h = bh & 15;
    __shared__ _Float16 Ks[192 * 64];
    __shared__ _Float16 Vs[192 * 64];
    __shared__ float qrow[64];
    __shared__ float sc[192];
    __shared__ float red[192];
    const int tid = threadIdx.x;
    const size_t base = ((size_t)b * T) * Dd + (size_t)h * HD;

    for (int i = tid; i < T * HD; i += 192) {
        int j = i >> 6, d = i & 63;
        Ks[i] = (_Float16)Kx[base + (size_t)j * Dd + d];
        Vs[i] = (_Float16)Vx[base + (size_t)j * Dd + d];
    }
    __syncthreads();

    for (int i = 0; i < T; ++i) {
        if (tid < HD) qrow[tid] = Q[base + (size_t)i * Dd + tid];
        __syncthreads();
        float s;
        if (tid <= i) {
            float acc = 0.f;
#pragma unroll 8
            for (int d = 0; d < HD; ++d)
                acc += qrow[d] * (float)Ks[tid * HD + d];
            s = acc * 0.125f;
        } else {
            s = -1e9f;
        }
        sc[tid] = s; red[tid] = s; __syncthreads();
        if (tid < 64) red[tid] = fmaxf(red[tid], red[tid + 128]); __syncthreads();
        for (int off = 64; off > 0; off >>= 1) {
            if (tid < off) red[tid] = fmaxf(red[tid], red[tid + off]);
            __syncthreads();
        }
        float mx = red[0]; __syncthreads();
        float e = expf(s - mx);
        sc[tid] = e; red[tid] = e; __syncthreads();
        if (tid < 64) red[tid] += red[tid + 128]; __syncthreads();
        for (int off = 64; off > 0; off >>= 1) {
            if (tid < off) red[tid] += red[tid + off];
            __syncthreads();
        }
        float inv = 1.0f / (1e-8f + red[0]); __syncthreads();
        sc[tid] = e * inv; __syncthreads();
        if (tid < HD) {
            float acc = 0.f;
            for (int j = 0; j <= i; ++j)
                acc += sc[j] * (float)Vs[j * HD + tid];
            Y[base + (size_t)i * Dd + tid] = acc;
        }
        __syncthreads();
    }
}

// ---------------------------------------------------------------------------
// rtg token: p@meta collapses to c1*bw + c0*bb. Writes X row 3l (+pos).
// ---------------------------------------------------------------------------
__global__ __launch_bounds__(128) void rtg_kernel(
    const float* __restrict__ rtgs, const float* __restrict__ ret_w,
    const float* __restrict__ res_w, const float* __restrict__ bw,
    const float* __restrict__ bb, const float* __restrict__ pos,
    float* __restrict__ X)
{
    const int NBK = 100, Dd = 1024, T = 192, L = 64;
    const int tok = blockIdx.x, b = tok / L, l = tok % L;
    const int tid = threadIdx.x;
    __shared__ float s[100], e[100], coef[2];
    const float rtg = rtgs[tok];
    if (tid < NBK) {
        float t = rtg * ret_w[tid];
        s[tid] = (t > 0.f) ? t : 0.01f * t;
    }
    __syncthreads();
    if (tid < NBK) {
        float acc = s[tid];
        for (int i = 0; i < NBK; ++i) acc += s[i] * res_w[i * NBK + tid];
        e[tid] = acc;
    }
    __syncthreads();
    if (tid == 0) {
        float mx = -1e30f;
        for (int j = 0; j < NBK; ++j) mx = fmaxf(mx, e[j]);
        float sum = 0.f;
        for (int j = 0; j < NBK; ++j) { float ev = expf(e[j] - mx); e[j] = ev; sum += ev; }
        float inv = 1.0f / (1e-8f + sum);
        float c0 = 0.f, c1 = 0.f;
        for (int j = 0; j < NBK; ++j) {
            float p = e[j] * inv;
            c0 += p;
            c1 += p * (100.0f * (float)j / 99.0f);
        }
        coef[0] = c0; coef[1] = c1;
    }
    __syncthreads();
    const float c0 = coef[0], c1 = coef[1];
    const size_t row = (size_t)(b * T + 3 * l) * Dd;
    const size_t prow = (size_t)(3 * l) * Dd;
    for (int d = tid; d < Dd; d += 128)
        X[row + d] = c1 * bw[d] + c0 * bb[d] + pos[prow + d];
}

// ---------------------------------------------------------------------------
// Interleave state/action embeddings into X rows 3l+1, 3l+2 (+ pos_emb).
// ---------------------------------------------------------------------------
__global__ __launch_bounds__(256) void stack_kernel(
    const float* __restrict__ SE, const float* __restrict__ AE,
    const float* __restrict__ pos, float* __restrict__ X)
{
    const int Dd = 1024, T = 192, L = 64;
    const int tok = blockIdx.x, b = tok / L, l = tok % L;
    const size_t src  = (size_t)tok * Dd;
    const size_t dstS = (size_t)(b * T + 3 * l + 1) * Dd;
    const size_t dstA = (size_t)(b * T + 3 * l + 2) * Dd;
    const size_t pS = (size_t)(3 * l + 1) * Dd, pA = (size_t)(3 * l + 2) * Dd;
    for (int d = threadIdx.x; d < Dd; d += 256) {
        X[dstS + d] = SE[src + d] + pos[pS + d];
        X[dstA + d] = AE[src + d] + pos[pA + d];
    }
}

// ---------------------------------------------------------------------------
// Host-side orchestration
// ---------------------------------------------------------------------------
static inline void launch_gemm(int act, bool res_add,
                               const float* A, const float* W, const float* bias,
                               const float* res, float* C, int M, int N, int K,
                               hipStream_t stream)
{
    if ((M % FBM) == 0 && (N % FBN) == 0 && (K % FBK) == 0) {
        dim3 grid(N / FBN, M / FBM);
        if (act == 1)
            gemm_wmma_fast<1, false><<<grid, 256, 0, stream>>>(A, W, bias, nullptr, C, M, N, K);
        else if (res_add)
            gemm_wmma_fast<0, true><<<grid, 256, 0, stream>>>(A, W, bias, res, C, M, N, K);
        else
            gemm_wmma_fast<0, false><<<grid, 256, 0, stream>>>(A, W, bias, nullptr, C, M, N, K);
    } else {
        dim3 grid((N + BN - 1) / BN, (M + BM - 1) / BM);
        if (act == 1)
            gemm_wmma_edge<1, false><<<grid, 256, 0, stream>>>(A, W, bias, nullptr, C, M, N, K);
        else if (res_add)
            gemm_wmma_edge<0, true><<<grid, 256, 0, stream>>>(A, W, bias, res, C, M, N, K);
        else
            gemm_wmma_edge<0, false><<<grid, 256, 0, stream>>>(A, W, bias, nullptr, C, M, N, K);
    }
}

extern "C" void kernel_launch(void* const* d_in, const int* in_sizes, int n_in,
                              void* d_out, int out_size, void* d_ws, size_t ws_size,
                              hipStream_t stream)
{
    const int D = 1024, NL = 6, Bz = 16, L = 64, T = 3 * L, DF = 4 * D;
    const int BL = Bz * L;          // 1024 token rows for embeddings
    const int MT = Bz * T;          // 3072 sequence rows

    const float* states   = (const float*)d_in[0];
    const float* actions  = (const float*)d_in[1];
    const float* rtgs     = (const float*)d_in[2];
    const float* se_w1    = (const float*)d_in[5];
    const float* se_b1    = (const float*)d_in[6];
    const float* se_w2    = (const float*)d_in[7];
    const float* se_b2    = (const float*)d_in[8];
    const float* ae_w1    = (const float*)d_in[9];
    const float* ae_b1    = (const float*)d_in[10];
    const float* ae_w2    = (const float*)d_in[11];
    const float* ae_b2    = (const float*)d_in[12];
    const float* ad_bw    = (const float*)d_in[13];
    const float* ad_bb    = (const float*)d_in[14];
    const float* ad_ret_w = (const float*)d_in[15];
    const float* ad_res_w = (const float*)d_in[16];
    const float* pos_emb  = (const float*)d_in[17];
    const float* ln1_g    = (const float*)d_in[18];
    const float* ln1_b    = (const float*)d_in[19];
    const float* Wq       = (const float*)d_in[20];
    const float* bq       = (const float*)d_in[21];
    const float* Wk       = (const float*)d_in[22];
    const float* bk       = (const float*)d_in[23];
    const float* Wv       = (const float*)d_in[24];
    const float* bv       = (const float*)d_in[25];
    const float* Wp       = (const float*)d_in[26];
    const float* bp       = (const float*)d_in[27];
    const float* ln2_g    = (const float*)d_in[28];
    const float* ln2_b    = (const float*)d_in[29];
    const float* Wm1      = (const float*)d_in[30];
    const float* bm1      = (const float*)d_in[31];
    const float* Wm2      = (const float*)d_in[32];
    const float* bm2      = (const float*)d_in[33];
    const float* lin_w    = (const float*)d_in[34];
    const float* lin_b    = (const float*)d_in[35];

    // Workspace: 6 chunks of MT*D fp32 (~75 MB total).
    const size_t CH = (size_t)MT * D;
    float* X  = (float*)d_ws;
    float* Hb = X  + CH;
    float* Qb = Hb + CH;
    float* Kb = Qb + CH;
    float* Vb = Kb + CH;
    float* Yb = Vb + CH;
    float* M1 = Qb;                 // MLP intermediate spans Qb..Yb (MT*DF)

    // ---- embeddings ----
    launch_gemm(1, false, states, se_w1, se_b1, nullptr, Qb, BL, D, 91, stream);   // edge (K=91)
    launch_gemm(1, false, Qb, se_w2, se_b2, nullptr, Kb, BL, D, D, stream);        // fast
    launch_gemm(1, false, actions, ae_w1, ae_b1, nullptr, Vb, BL, D, 27, stream);  // edge (K=27)
    launch_gemm(0, false, Vb, ae_w2, ae_b2, nullptr, Yb, BL, D, D, stream);        // fast
    rtg_kernel<<<BL, 128, 0, stream>>>(rtgs, ad_ret_w, ad_res_w, ad_bw, ad_bb, pos_emb, X);
    stack_kernel<<<BL, 256, 0, stream>>>(Kb, Yb, pos_emb, X);

    // ---- transformer layers ----
    for (int i = 0; i < NL; ++i) {
        const size_t WD = (size_t)i * D * D;
        ln_kernel<<<MT, 256, 0, stream>>>(X, ln1_g + i * D, ln1_b + i * D, Hb, D);
        launch_gemm(0, false, Hb, Wq + WD, bq + i * D, nullptr, Qb, MT, D, D, stream);
        launch_gemm(0, false, Hb, Wk + WD, bk + i * D, nullptr, Kb, MT, D, D, stream);
        launch_gemm(0, false, Hb, Wv + WD, bv + i * D, nullptr, Vb, MT, D, D, stream);
        attn_kernel<<<Bz * 16, 192, 0, stream>>>(Qb, Kb, Vb, Yb);
        launch_gemm(0, true, Yb, Wp + WD, bp + i * D, X, X, MT, D, D, stream);
        ln_kernel<<<MT, 256, 0, stream>>>(X, ln2_g + i * D, ln2_b + i * D, Hb, D);
        launch_gemm(1, false, Hb, Wm1 + (size_t)i * D * DF, bm1 + i * DF, nullptr, M1, MT, DF, D, stream);
        launch_gemm(0, true, M1, Wm2 + (size_t)i * DF * D, bm2 + i * D, X, X, MT, D, DF, stream);
    }

    // ---- head: (3072x1024) @ (1024x27) + bias -> d_out (edge, N=27) ----
    launch_gemm(0, false, X, lin_w, lin_b, nullptr, (float*)d_out, MT, 27, D, stream);
}
